// RvNN_TD_All_Nodes_84911503442504
// MI455X (gfx1250) — matrix operand order repaired and checked by reading, not compiled, not run on hardware
//
#include <hip/hip_runtime.h>
#include <hip/hip_bf16.h>
#include <math.h>

// ---------------------------------------------------------------------------
// RvNN top-down tree-GRU for MI455X (gfx1250, wave32, WMMA).
//   B=64, N=256, F_IN=768, F_OUT=256.
//   * All WMMA operands pre-packed bf16 in memory; fragments load as two
//     16-byte packed loads (no per-element conversion ALU).
//   * Scan: one workgroup per batch; bf16 h_copy (feature-major, 128KB) in
//     CDNA5's 320KB LDS; blocked 16-node aggregation via WMMA + exact VALU
//     intra-block correction; f32 GEMV recurrence.
// ---------------------------------------------------------------------------

typedef __attribute__((ext_vector_type(16))) __bf16 v16bf;
typedef __attribute__((ext_vector_type(8)))  __bf16 bf16x8;
typedef __attribute__((ext_vector_type(8)))  float  v8f;

#define BATCH 64
#define NNODE 256
#define FIN   768
#define FOUT  256
#define BLK   16

// float -> bf16 (round-to-nearest-even) via bit manipulation (no fp convert).
__device__ __forceinline__ __bf16 tobf(float f) {
    union { float f; unsigned u; } in; in.f = f;
    unsigned u = in.u;
    unsigned r = (u + 0x7FFFu + ((u >> 16) & 1u)) >> 16;
    union { unsigned short s; __bf16 b; } out;
    out.s = (unsigned short)r;
    return out.b;
}
__device__ __forceinline__ float bf2f(__bf16 b) {
    union { unsigned short s; __bf16 b; } in; in.b = b;
    union { unsigned u; float f; } out; out.u = ((unsigned)in.s) << 16;
    return out.f;
}
__device__ __forceinline__ v16bf concat8(bf16x8 lo, bf16x8 hi) {
    return __builtin_shufflevector(lo, hi, 0,1,2,3,4,5,6,7,8,9,10,11,12,13,14,15);
}

// A fragment (16x32 bf16) from row-major bf16 src (ld multiple of 8).
// Lane L: row M=L%16; element e -> K = (e%8) + 16*(e/8) + 8*(L/16).
// => two contiguous 8xbf16 (16B) runs at k0+khi and k0+khi+16.
__device__ __forceinline__ v16bf load_a_frag(const __bf16* src, int ld, int m0, int k0, int lane) {
    const int m   = m0 + (lane & 15);
    const int khi = ((lane >> 4) << 3);
    const __bf16* p = src + (size_t)m * ld + k0 + khi;
    return concat8(*(const bf16x8*)p, *(const bf16x8*)(p + 16));
}

// B fragment (32x16 bf16) with B[k,n] = src[n*ld + k] (src row-major [N x K]).
// Lane L: col N=L%16; element e -> K = (e%8) + 16*(e/8) + 8*(L/16) => two 8xbf16 runs.
__device__ __forceinline__ v16bf load_b_frag(const __bf16* src, int ld, int n0, int k0, int lane) {
    const int n  = n0 + (lane & 15);
    const int kb = k0 + ((lane >> 4) << 3);
    const __bf16* p = src + (size_t)n * ld + kb;
    return concat8(*(const bf16x8*)p, *(const bf16x8*)(p + 16));
}

// ---------------------------------------------------------------------------
// f32 -> bf16 pack (bandwidth-bound, one-shot)
// ---------------------------------------------------------------------------
__global__ void f32_to_bf16_kernel(const float* __restrict__ src, __bf16* __restrict__ dst, int n) {
    int i = blockIdx.x * blockDim.x + threadIdx.x;
    if (i < n) dst[i] = tobf(src[i]);
}

// ---------------------------------------------------------------------------
// C[M,N] = A[M,K] @ W[N,K]^T (+ bias); A,W bf16; one wave owns a 32x64 tile.
// Optional f32 and/or bf16 outputs.
// ---------------------------------------------------------------------------
__global__ void wmma_gemm_nt(const __bf16* __restrict__ A, const __bf16* __restrict__ W,
                             const float* __restrict__ bias,
                             float* __restrict__ Cf, __bf16* __restrict__ Cb,
                             int M, int K, int N) {
    const int lane   = threadIdx.x & 31;
    const int waveId = (int)((blockIdx.x * blockDim.x + threadIdx.x) >> 5);
    const int mTiles = M >> 5;                 // 32-row tiles
    const int m0 = (waveId % mTiles) << 5;
    const int n0 = (waveId / mTiles) << 6;     // 64 columns per wave
    if (n0 >= N) return;

    v8f acc0[4] = {};
    v8f acc1[4] = {};
    for (int k0 = 0; k0 < K; k0 += 32) {
        v16bf a0 = load_a_frag(A, K, m0,      k0, lane);
        v16bf a1 = load_a_frag(A, K, m0 + 16, k0, lane);
#pragma unroll
        for (int t = 0; t < 4; ++t) {
            v16bf b = load_b_frag(W, K, n0 + t * 16, k0, lane);
            acc0[t] = __builtin_amdgcn_wmma_f32_16x16x32_bf16(
                false, a0, false, b, (short)0, acc0[t], false, false);
            acc1[t] = __builtin_amdgcn_wmma_f32_16x16x32_bf16(
                false, a1, false, b, (short)0, acc1[t], false, false);
        }
    }
    const int mrow  = m0 + ((lane >> 4) << 3);  // + r gives M (first 16-row tile)
    const int ncol0 = n0 + (lane & 15);
#pragma unroll
    for (int t = 0; t < 4; ++t) {
        const int n = ncol0 + t * 16;
        const float bv = bias ? bias[n] : 0.0f;
#pragma unroll
        for (int r = 0; r < 8; ++r) {
            float v0 = acc0[t][r] + bv;
            float v1 = acc1[t][r] + bv;
            if (Cf) {
                Cf[(size_t)(mrow + r) * N + n]      = v0;
                Cf[(size_t)(mrow + 16 + r) * N + n] = v1;
            }
            if (Cb) {
                Cb[(size_t)(mrow + r) * N + n]      = tobf(v0);
                Cb[(size_t)(mrow + 16 + r) * N + n] = tobf(v1);
            }
        }
    }
}

// WT[k*O + o] = W[o*K + k]
__global__ void transpose_kernel(const float* __restrict__ W, float* __restrict__ WT,
                                 int O, int K) {
    int idx = blockIdx.x * blockDim.x + threadIdx.x;
    if (idx >= O * K) return;
    int o = idx / K, k = idx - o * K;
    WT[(size_t)k * O + o] = W[(size_t)o * K + k];
}

// ---------------------------------------------------------------------------
// Sequential tree-GRU scan: one workgroup (256 threads, 8 waves) per batch.
// h_copy kept bf16 feature-major in LDS; aggregation per 16-node block via
// WMMA; exact intra-block correction + f32 GEMVs per step.
// ---------------------------------------------------------------------------
__global__ void scan_kernel(const float*  __restrict__ adj,     // [B,N,N] f32
                            const __bf16* __restrict__ heb,     // [B,N,F] bf16
                            const float*  __restrict__ ioux,    // [B,N,2F] (incl. ioux_b)
                            const float*  __restrict__ coux,    // [B,N,F]  (incl. coux_b)
                            const float*  __restrict__ iouhWT,  // [F, 2F]
                            const float*  __restrict__ iouh_b,  // [2F]
                            const float*  __restrict__ couhWT,  // [F, F]
                            const float*  __restrict__ couh_b,  // [F]
                            float* __restrict__ pooled)         // [B,F]
{
    extern __shared__ __align__(16) char smraw[];
    float*  sm_S     = (float*)smraw;                        // 16*256 f32   (16384B)
    float*  sm_delta = sm_S     + BLK * FOUT;                // 16*256 f32   (16384B)
    float*  sm_dinv  = sm_delta + BLK * FOUT;                // 256 f32
    float*  sm_x     = sm_dinv  + NNODE;                     // 256 f32
    float*  sm_rx    = sm_x     + FOUT;                      // 256 f32
    float*  sm_z     = sm_rx    + FOUT;                      // 256 f32
    __bf16* sm_hT    = (__bf16*)(sm_z + FOUT);               // [F][N] bf16 (131072B)
    __bf16* sm_aT    = sm_hT + (size_t)FOUT * NNODE;         // [16][N] bf16 (8192B)

    const int b    = blockIdx.x;
    const int tid  = threadIdx.x;            // blockDim == 256 == FOUT == NNODE
    const int lane = tid & 31;
    const int wave = tid >> 5;
    const float* adj_b = adj + (size_t)b * NNODE * NNODE;

    // h_copy (transposed, bf16): sm_hT[f][j] = h_e[b, j, f]
    for (int idx = tid; idx < NNODE * FOUT; idx += 256) {
        int j = idx >> 8, f = idx & 255;
        sm_hT[(size_t)f * NNODE + j] = heb[(size_t)b * NNODE * FOUT + idx];
    }
    // d_inv[n] = 1 / row-degree (0 if degree 0)
    {
        float d = 0.f;
        const float* row = adj_b + (size_t)tid * NNODE;
        for (int j = 0; j < NNODE; ++j) d += row[j];
        sm_dinv[tid] = (d != 0.f) ? 1.f / d : 0.f;
    }
    __syncthreads();

    for (int i0 = 0; i0 < NNODE; i0 += BLK) {
        // Stage adj^T block as bf16 (0/1 exact): aT[di][j] = adj[b, j, i0+di]
        for (int idx = tid; idx < BLK * NNODE; idx += 256) {
            int di = idx >> 8, j = idx & 255;
            sm_aT[idx] = tobf(adj_b[(size_t)j * NNODE + (i0 + di)]);
        }
        __syncthreads();

        // WMMA: S[16 x 256] = aT(16 x 256_j) * h(j x f); B[k=j, n=f] = sm_hT[f][j]
        {
            const int n0 = wave * 32;   // 8 waves cover 256 features
            v8f acc0 = {}, acc1 = {};
            for (int kc = 0; kc < NNODE; kc += 32) {
                v16bf a  = load_a_frag(sm_aT, NNODE, 0, kc, lane);
                v16bf b0 = load_b_frag(sm_hT, NNODE, n0,      kc, lane);
                v16bf b1 = load_b_frag(sm_hT, NNODE, n0 + 16, kc, lane);
                acc0 = __builtin_amdgcn_wmma_f32_16x16x32_bf16(
                    false, a, false, b0, (short)0, acc0, false, false);
                acc1 = __builtin_amdgcn_wmma_f32_16x16x32_bf16(
                    false, a, false, b1, (short)0, acc1, false, false);
            }
            const int mrow = ((lane >> 4) << 3);
            const int n    = lane & 15;
#pragma unroll
            for (int r = 0; r < 8; ++r) {
                sm_S[(mrow + r) * FOUT + n0 + n]      = acc0[r];
                sm_S[(mrow + r) * FOUT + n0 + 16 + n] = acc1[r];
            }
        }
        __syncthreads();

        // 16 sequential GRU steps; thread tid owns feature f = tid.
        for (int di = 0; di < BLK; ++di) {
            const int i = i0 + di;
            const int f = tid;

            // x_parent with exact intra-block correction
            float corr = 0.f;
            for (int j = 0; j < di; ++j)
                corr += bf2f(sm_aT[di * NNODE + (i0 + j)]) * sm_delta[j * FOUT + f];
            const float x = sm_dinv[i] * (sm_S[di * FOUT + f] + corr);
            sm_x[f] = x;
            __syncthreads();

            // iou = iou_x[b,i] + iouh_W @ x + iouh_b ; outputs f and f+256
            float a0 = ioux[((size_t)b * NNODE + i) * (2 * FOUT) + f]        + iouh_b[f];
            float a1 = ioux[((size_t)b * NNODE + i) * (2 * FOUT) + f + FOUT] + iouh_b[f + FOUT];
#pragma unroll 4
            for (int k = 0; k < FOUT; ++k) {
                const float xv = sm_x[k];
                a0 += iouhWT[(size_t)k * (2 * FOUT) + f]        * xv;
                a1 += iouhWT[(size_t)k * (2 * FOUT) + f + FOUT] * xv;
            }
            const float r_ = 1.f / (1.f + __expf(-a0));
            const float z_ = 1.f / (1.f + __expf(-a1));
            sm_rx[f] = r_ * x;
            sm_z[f]  = z_;
            __syncthreads();

            // hc = tanh(cou_x[b,i] + couh_W @ (r*x) + couh_b)
            float c0 = coux[((size_t)b * NNODE + i) * FOUT + f] + couh_b[f];
#pragma unroll 4
            for (int k = 0; k < FOUT; ++k)
                c0 += couhWT[(size_t)k * FOUT + f] * sm_rx[k];
            const float hc = tanhf(c0);
            const float z  = sm_z[f];
            const float hn = z * sm_x[f] + (1.f - z) * hc;

            const float hold = bf2f(sm_hT[(size_t)f * NNODE + i]);
            sm_delta[di * FOUT + f]        = hn - hold;
            sm_hT[(size_t)f * NNODE + i]   = tobf(hn);
            __syncthreads();
        }
    }

    // Max-pool over nodes (contiguous per thread: sm_hT[f][0..N))
    {
        const int f = tid;
        const __bf16* row = sm_hT + (size_t)f * NNODE;
        float m = bf2f(row[0]);
        for (int n = 1; n < NNODE; ++n) m = fmaxf(m, bf2f(row[n]));
        pooled[(size_t)b * FOUT + f] = m;
    }
}

// out[b,c] = pooled[b] . fc_W[c] + fc_b[c]
__global__ void fc_kernel(const float* __restrict__ pooled, const float* __restrict__ fcW,
                          const float* __restrict__ fcb, float* __restrict__ out) {
    int t = blockIdx.x * blockDim.x + threadIdx.x;
    if (t >= BATCH * 2) return;
    int b = t >> 1, c = t & 1;
    float acc = fcb[c];
    for (int f = 0; f < FOUT; ++f)
        acc += pooled[(size_t)b * FOUT + f] * fcW[(size_t)c * FOUT + f];
    out[t] = acc;
}

extern "C" void kernel_launch(void* const* d_in, const int* in_sizes, int n_in,
                              void* d_out, int out_size, void* d_ws, size_t ws_size,
                              hipStream_t stream) {
    const float* h      = (const float*)d_in[0];
    const float* adj    = (const float*)d_in[1];
    const float* emb_W  = (const float*)d_in[2];
    const float* ioux_W = (const float*)d_in[3];
    const float* ioux_b = (const float*)d_in[4];
    const float* iouh_W = (const float*)d_in[5];
    const float* iouh_b = (const float*)d_in[6];
    const float* coux_W = (const float*)d_in[7];
    const float* coux_b = (const float*)d_in[8];
    const float* couh_W = (const float*)d_in[9];
    const float* couh_b = (const float*)d_in[10];
    const float* fc_W   = (const float*)d_in[11];
    const float* fc_b   = (const float*)d_in[12];
    float* out = (float*)d_out;

    char* ws = (char*)d_ws;
    float*  ws_ioux   = (float*)ws;  ws += (size_t)BATCH * NNODE * 2 * FOUT * 4;
    float*  ws_coux   = (float*)ws;  ws += (size_t)BATCH * NNODE * FOUT * 4;
    float*  ws_iouhWT = (float*)ws;  ws += (size_t)2 * FOUT * FOUT * 4;
    float*  ws_couhWT = (float*)ws;  ws += (size_t)FOUT * FOUT * 4;
    float*  ws_pool   = (float*)ws;  ws += (size_t)BATCH * FOUT * 4;
    __bf16* ws_hb     = (__bf16*)ws; ws += (size_t)BATCH * NNODE * FIN * 2;
    __bf16* ws_heb    = (__bf16*)ws; ws += (size_t)BATCH * NNODE * FOUT * 2;
    __bf16* ws_embWb  = (__bf16*)ws; ws += (size_t)FOUT * FIN * 2;
    __bf16* ws_iouxWb = (__bf16*)ws; ws += (size_t)2 * FOUT * FOUT * 2;
    __bf16* ws_couxWb = (__bf16*)ws; ws += (size_t)FOUT * FOUT * 2;

    const int M = BATCH * NNODE;  // 16384 rows

    // --- pack WMMA operands to bf16 (bandwidth-bound) ---
    {
        int n;
        n = M * FIN;            f32_to_bf16_kernel<<<(n + 255) / 256, 256, 0, stream>>>(h,      ws_hb,     n);
        n = FOUT * FIN;         f32_to_bf16_kernel<<<(n + 255) / 256, 256, 0, stream>>>(emb_W,  ws_embWb,  n);
        n = 2 * FOUT * FOUT;    f32_to_bf16_kernel<<<(n + 255) / 256, 256, 0, stream>>>(ioux_W, ws_iouxWb, n);
        n = FOUT * FOUT;        f32_to_bf16_kernel<<<(n + 255) / 256, 256, 0, stream>>>(coux_W, ws_couxWb, n);
    }

    // h_e = h @ emb_W^T  -> bf16 only (consumed by GEMMs + scan)
    {
        int waves = (M / 32) * (FOUT / 64);
        wmma_gemm_nt<<<waves / 8, 256, 0, stream>>>(ws_hb, ws_embWb, nullptr,
                                                    nullptr, ws_heb, M, FIN, FOUT);
    }
    // iou_x = h_e @ ioux_W^T + ioux_b  -> f32
    {
        int waves = (M / 32) * ((2 * FOUT) / 64);
        wmma_gemm_nt<<<waves / 8, 256, 0, stream>>>(ws_heb, ws_iouxWb, ioux_b,
                                                    ws_ioux, nullptr, M, FOUT, 2 * FOUT);
    }
    // cou_x = h_e @ coux_W^T + coux_b  -> f32
    {
        int waves = (M / 32) * (FOUT / 64);
        wmma_gemm_nt<<<waves / 8, 256, 0, stream>>>(ws_heb, ws_couxWb, coux_b,
                                                    ws_coux, nullptr, M, FOUT, FOUT);
    }
    // Lane-coalesced transposed recurrence weights (f32, read from L2 each step)
    transpose_kernel<<<(2 * FOUT * FOUT + 255) / 256, 256, 0, stream>>>(iouh_W, ws_iouhWT, 2 * FOUT, FOUT);
    transpose_kernel<<<(FOUT * FOUT + 255) / 256, 256, 0, stream>>>(couh_W, ws_couhWT, FOUT, FOUT);

    // Scan: one workgroup per batch; ~176KB LDS
    size_t smem = (size_t)(2 * BLK * FOUT + NNODE + 3 * FOUT) * sizeof(float)
                + (size_t)(FOUT * NNODE + BLK * NNODE) * sizeof(__bf16);
    scan_kernel<<<BATCH, 256, smem, stream>>>(adj, ws_heb, ws_ioux, ws_coux,
                                              ws_iouhWT, iouh_b, ws_couhWT, couh_b, ws_pool);

    fc_kernel<<<1, 128, 0, stream>>>(ws_pool, fc_W, fc_b, out);
}
